// GCNEncoder_65910568124789
// MI455X (gfx1250) — compile-verified, matrix operand order checked by reading, not run on hardware
//
#include <hip/hip_runtime.h>

// ---------------------------------------------------------------------------
// GCN encoder forward for MI455X (gfx1250, wave32, WMMA bf16, fp32 accum).
// Workspace layout (floats):
//   m[N*128] | z[N*128] | deg_scales[2*N] (out|in) | stats[256] | Wt_bf16[16384]
// d_out doubles as h1 storage between layers (residual read+overwrite in place).
// ---------------------------------------------------------------------------

typedef __attribute__((ext_vector_type(16))) __bf16 v16bf;
typedef __attribute__((ext_vector_type(8)))  __bf16 v8bf;
typedef __attribute__((ext_vector_type(8)))  float  v8f;

#define WS 136  // LDS row stride in bf16 elems (mult. of 8 -> 16B-aligned chunks)

__global__ void zero_f(float* __restrict__ p, long long n) {
  long long g = (long long)blockIdx.x * blockDim.x + threadIdx.x;
  if (g < n) p[g] = 0.f;
}

__global__ void deg_accum(const int* __restrict__ src, const int* __restrict__ dst,
                          float* __restrict__ outdeg, float* __restrict__ indeg, int E) {
  int g = blockIdx.x * blockDim.x + threadIdx.x;
  if (g < E) {
    atomicAdd(&outdeg[src[g]], 1.f);
    atomicAdd(&indeg[dst[g]], 1.f);
  }
}

__global__ void deg_finalize(float* __restrict__ deg, long long n) {
  long long g = (long long)blockIdx.x * blockDim.x + threadIdx.x;
  if (g < n) deg[g] = rsqrtf(fmaxf(deg[g], 1.f));  // clip(deg,1)^-0.5
}

// Transposed bf16 copy of W: Wt[n*128 + k] = bf16(W[k*128 + n])
__global__ void convert_w(const float* __restrict__ W, __bf16* __restrict__ Wt) {
  int i = blockIdx.x * blockDim.x + threadIdx.x;  // 16384 threads
  int k = i >> 7, n = i & 127;
  Wt[n * 128 + k] = (__bf16)W[i];
}

// m[dst] += h[src] * d_out_scale[src] ; one wave per edge, float4 per lane
__global__ void aggregate(const float* __restrict__ h, const float* __restrict__ dsc,
                          const int* __restrict__ src, const int* __restrict__ dst,
                          float* __restrict__ m, long long E) {
  long long gid = (long long)blockIdx.x * blockDim.x + threadIdx.x;
  if (gid >= E * 32) return;
  long long e = gid >> 5;
  int q = (int)(gid & 31);
  int s = src[e], d = dst[e];
  float sc = dsc[s];
  float4 v = ((const float4*)(h + (size_t)s * 128))[q];
  float* base = m + (size_t)d * 128 + q * 4;
  atomicAdd(base + 0, v.x * sc);
  atomicAdd(base + 1, v.y * sc);
  atomicAdd(base + 2, v.z * sc);
  atomicAdd(base + 3, v.w * sc);
}

// z = diag(d_in) * (m @ W) + b   via v_wmma_f32_16x16x32_bf16.
// Block: 256 thr = 8 wave32; tile = 16 rows x 128 cols; K = 128 (4 WMMAs/wave).
// A tile staged bf16 in LDS (8x reuse across waves); B fragments read straight
// from the pre-transposed bf16 W in global (L1/L2 resident, no reuse in-block).
__global__ __launch_bounds__(256)
void gemm_wmma(const float* __restrict__ m, const float* __restrict__ din,
               const __bf16* __restrict__ Wt, const float* __restrict__ bias,
               float* __restrict__ z, int N) {
  __shared__ __align__(16) __bf16 sA[16 * WS];
  __shared__ float sDin[16];

  const int tid  = threadIdx.x;
  const int row0 = blockIdx.x * 16;

  // stage A tile fp32 -> bf16 (512 float4 chunks)
  for (int i = tid; i < 512; i += 256) {
    int r = i >> 5, c = (i & 31) * 4;
    int row = row0 + r;
    float4 v = make_float4(0.f, 0.f, 0.f, 0.f);
    if (row < N) v = *(const float4*)(m + (size_t)row * 128 + c);
    __bf16* p = &sA[r * WS + c];
    p[0] = (__bf16)v.x; p[1] = (__bf16)v.y;
    p[2] = (__bf16)v.z; p[3] = (__bf16)v.w;
  }
  if (tid < 16) {
    int row = row0 + tid;
    sDin[tid] = (row < N) ? din[row] : 0.f;
  }
  __syncthreads();

  const int wave = tid >> 5;
  const int lane = tid & 31;
  const int half = lane >> 4;  // K-half selector per ISA 16-bit layout
  const int idx  = lane & 15;  // M (A fragment) / N (B fragment)
  const int col0 = wave * 16;

  const __bf16* wb = Wt + (size_t)(col0 + idx) * 128;  // column (col0+idx) of W

  v8f acc = {};
#pragma unroll
  for (int kt = 0; kt < 4; ++kt) {
    const int k0 = kt * 32;
    const v8bf a_lo = *(const v8bf*)&sA[idx * WS + k0 + half * 8];
    const v8bf a_hi = *(const v8bf*)&sA[idx * WS + k0 + 16 + half * 8];
    const v8bf b_lo = *(const v8bf*)(wb + k0 + half * 8);
    const v8bf b_hi = *(const v8bf*)(wb + k0 + 16 + half * 8);
    v16bf a, b;
#pragma unroll
    for (int i = 0; i < 8; ++i) {
      a[i] = a_lo[i]; a[i + 8] = a_hi[i];
      b[i] = b_lo[i]; b[i + 8] = b_hi[i];
    }
    acc = __builtin_amdgcn_wmma_f32_16x16x32_bf16(
        /*neg_a=*/false, a, /*neg_b=*/false, b,
        /*c_mod=*/(short)0, acc, /*reuse_a=*/false, /*reuse_b=*/false);
  }

  // epilogue: row = row0 + 8*half + r, col = col0 + idx
  const float bc = bias[col0 + idx];
  float* zp = z + (size_t)(row0 + half * 8) * 128 + col0 + idx;
  if (row0 + 16 <= N) {  // uniform fast path: branch-free stores
#pragma unroll
    for (int r = 0; r < 8; ++r)
      zp[(size_t)r * 128] = acc[r] * sDin[half * 8 + r] + bc;
  } else {
#pragma unroll
    for (int r = 0; r < 8; ++r)
      if (row0 + half * 8 + r < N)
        zp[(size_t)r * 128] = acc[r] * sDin[half * 8 + r] + bc;
  }
}

// per-column sum / sumsq partials -> float atomics
__global__ void col_stats(const float* __restrict__ z, float* __restrict__ colsum,
                          float* __restrict__ colsq, int N) {
  int c = threadIdx.x;  // 128 threads = 128 columns, coalesced rows
  float s = 0.f, q = 0.f;
  for (int r = blockIdx.x; r < N; r += gridDim.x) {
    float v = z[(size_t)r * 128 + c];
    s += v;
    q += v * v;
  }
  atomicAdd(&colsum[c], s);
  atomicAdd(&colsq[c], q);
}

// out = relu(gamma*(z-mu)*rsqrt(var+eps)+beta [+ resid])
__global__ void bn_apply(const float* __restrict__ z, const float* __restrict__ colsum,
                         const float* __restrict__ colsq, const float* __restrict__ gamma,
                         const float* __restrict__ beta, float* __restrict__ out,
                         const float* __restrict__ resid, int N, int add_resid) {
  long long gid = (long long)blockIdx.x * blockDim.x + threadIdx.x;
  if (gid >= (long long)N * 128) return;
  int c = (int)(gid & 127);
  float inv = 1.f / (float)N;
  float mu = colsum[c] * inv;
  float var = colsq[c] * inv - mu * mu;
  float rs = rsqrtf(var + 1e-5f);
  float v = gamma[c] * (z[gid] - mu) * rs + beta[c];
  if (add_resid) v += resid[gid];
  out[gid] = fmaxf(v, 0.f);
}

static inline int cdiv(long long a, long long b) { return (int)((a + b - 1) / b); }

extern "C" void kernel_launch(void* const* d_in, const int* in_sizes, int n_in,
                              void* d_out, int out_size, void* d_ws, size_t ws_size,
                              hipStream_t stream) {
  const float* x   = (const float*)d_in[0];
  const int*   src = (const int*)d_in[1];
  const int*   dst = (const int*)d_in[2];
  const float* W1  = (const float*)d_in[3];
  const float* b1  = (const float*)d_in[4];
  const float* g1  = (const float*)d_in[5];
  const float* be1 = (const float*)d_in[6];
  const float* W2  = (const float*)d_in[7];
  const float* b2  = (const float*)d_in[8];
  const float* g2  = (const float*)d_in[9];
  const float* be2 = (const float*)d_in[10];

  const int N = in_sizes[0] / 128;
  const long long E = in_sizes[1];
  float* out = (float*)d_out;

  float* m     = (float*)d_ws;
  float* zb    = m + (size_t)N * 128;
  float* degs  = zb + (size_t)N * 128;     // [0,N): out scale, [N,2N): in scale
  float* stats = degs + (size_t)2 * N;     // 128 colsum + 128 colsumsq
  __bf16* Wt   = (__bf16*)(stats + 256);   // 128x128 transposed bf16 W

  // degree norms
  zero_f<<<cdiv(2LL * N + 256, 256), 256, 0, stream>>>(degs, 2LL * N + 256);
  deg_accum<<<cdiv(E, 256), 256, 0, stream>>>(src, dst, degs, degs + N, (int)E);
  deg_finalize<<<cdiv(2LL * N, 256), 256, 0, stream>>>(degs, 2LL * N);

  // ---- layer 1 ----
  zero_f<<<cdiv((long long)N * 128, 256), 256, 0, stream>>>(m, (long long)N * 128);
  aggregate<<<cdiv(E * 32, 256), 256, 0, stream>>>(x, degs, src, dst, m, E);
  convert_w<<<64, 256, 0, stream>>>(W1, Wt);
  gemm_wmma<<<cdiv(N, 16), 256, 0, stream>>>(m, degs + N, Wt, b1, zb, N);
  zero_f<<<1, 256, 0, stream>>>(stats, 256);
  col_stats<<<512, 128, 0, stream>>>(zb, stats, stats + 128, N);
  bn_apply<<<cdiv((long long)N * 128, 256), 256, 0, stream>>>(
      zb, stats, stats + 128, g1, be1, out, nullptr, N, 0);  // h1 -> d_out

  // ---- layer 2 ----
  zero_f<<<cdiv((long long)N * 128, 256), 256, 0, stream>>>(m, (long long)N * 128);
  aggregate<<<cdiv(E * 32, 256), 256, 0, stream>>>(out, degs, src, dst, m, E);
  convert_w<<<64, 256, 0, stream>>>(W2, Wt);
  gemm_wmma<<<cdiv(N, 16), 256, 0, stream>>>(m, degs + N, Wt, b2, zb, N);
  zero_f<<<1, 256, 0, stream>>>(stats, 256);
  col_stats<<<512, 128, 0, stream>>>(zb, stats, stats + 128, N);
  bn_apply<<<cdiv((long long)N * 128, 256), 256, 0, stream>>>(
      zb, stats, stats + 128, g2, be2, out, out, N, 1);  // out = relu(bn(z)+h1)
}
